// NWC_vq_69518340653322
// MI455X (gfx1250) — matrix operand order repaired and checked by reading, not compiled, not run on hardware
//
#include <hip/hip_runtime.h>

// ---------------------------------------------------------------------------
// VQ-VAE forward for MI455X (gfx1250, wave32, WMMA).
// GEMMs: bf16 x bf16 -> f32 via v_wmma_f32_16x16x32_bf16, with
// global->LDS staging done by GLOBAL_LOAD_ASYNC_TO_LDS_B128 (ASYNCcnt),
// double-buffered in LDS so DMA overlaps WMMA.
// ---------------------------------------------------------------------------

typedef __attribute__((ext_vector_type(16))) __bf16 v16bf;
typedef __attribute__((ext_vector_type(8)))  float  v8f;
typedef __bf16 bf16;

#define NN   32768
#define DIN  128
#define DH   1024
#define MMDIM 128
#define EDIM 16
#define KCODE 512
#define NROWS_VQ (NN * MMDIM / EDIM)   // 262144

#define BM 128
#define BN 128
#define BKK 32

// LDS aperture: generic address low 32 bits == LDS byte address (ISA 10.2).
__device__ __forceinline__ unsigned lds_lo(const void* p) {
    return (unsigned)(unsigned long long)p;
}

// Per-lane async DMA: 16 bytes global -> LDS, tracked by ASYNCcnt.
__device__ __forceinline__ void async_b128(unsigned lds_addr, const void* gaddr) {
    asm volatile("global_load_async_to_lds_b128 %0, %1, off"
                 :: "v"(lds_addr), "v"(gaddr) : "memory");
}

// ---------------------------------------------------------------------------
// GEMM: C[M,Nc] = A[M,K] @ B[K,Nc] + bias   (A,B bf16 row-major, C f32/bf16)
// 256 threads = 8 waves; each wave computes a 32x64 sub-tile (2x4 WMMA tiles).
// ---------------------------------------------------------------------------
__global__ __launch_bounds__(256)
void gemm_bf16_wmma_kernel(const bf16* __restrict__ A, const bf16* __restrict__ B,
                           const float* __restrict__ bias,
                           float* __restrict__ Cf, bf16* __restrict__ Cb,
                           int M, int Nc, int K)
{
    __shared__ alignas(32) bf16 As[2][BM * BKK];   // 2 x 8 KB
    __shared__ alignas(32) bf16 Bs[2][BKK * BN];   // 2 x 8 KB

    const int tid  = threadIdx.x;
    const int lane = tid & 31;
    const int wave = tid >> 5;          // 0..7
    const int wr   = wave >> 1;         // 0..3 (row of waves)
    const int wc   = wave & 1;          // 0..1 (col of waves)

    const int row0 = blockIdx.y * BM;
    const int col0 = blockIdx.x * BN;

    const unsigned asb[2] = { lds_lo(As[0]), lds_lo(As[1]) };
    const unsigned bsb[2] = { lds_lo(Bs[0]), lds_lo(Bs[1]) };

    v8f acc[2][4];
#pragma unroll
    for (int i = 0; i < 2; ++i)
#pragma unroll
        for (int j = 0; j < 4; ++j)
#pragma unroll
            for (int r = 0; r < 8; ++r) acc[i][j][r] = 0.0f;

    // Issue one k-stage (A tile 128x32, B tile 32x128) as 4 async b128/thread.
    auto issue = [&](int buf, int k0) {
#pragma unroll
        for (int it = 0; it < 2; ++it) {            // A: 512 chunks of 8 bf16
            int c  = tid + it * 256;
            int r  = c >> 2;                        // 4 chunks per row
            int ko = (c & 3) << 3;
            async_b128(asb[buf] + (unsigned)((r * BKK + ko) * 2),
                       A + (size_t)(row0 + r) * K + (k0 + ko));
        }
#pragma unroll
        for (int it = 0; it < 2; ++it) {            // B: 512 chunks of 8 bf16
            int c  = tid + it * 256;
            int r  = c >> 4;                        // 16 chunks per row
            int no = (c & 15) << 3;
            async_b128(bsb[buf] + (unsigned)((r * BN + no) * 2),
                       B + (size_t)(k0 + r) * Nc + (col0 + no));
        }
    };

    const int nst = K / BKK;
    issue(0, 0);

    for (int s = 0; s < nst; ++s) {
        const int cur = s & 1;
        __syncthreads();                 // all waves done computing stage s-1:
                                         // safe to overwrite buf cur^1
        if (s + 1 < nst) {
            issue(cur ^ 1, (s + 1) * BKK);
            asm volatile("s_wait_asynccnt 4" ::: "memory");   // stage s complete
        } else {
            asm volatile("s_wait_asynccnt 0" ::: "memory");
        }
        __syncthreads();                 // stage s data visible to every wave

        const bf16* Asc = As[cur];
        const bf16* Bsc = Bs[cur];

        // A fragments (ISA 16-bit A 16x32 layout: lane<16 -> K 0..7,16..23;
        // lane>=16 -> K 8..15,24..31)
        v16bf afrag[2];
        {
            const int arow = lane & 15;
            const int kh   = (lane >> 4) << 3;  // 0 or 8
#pragma unroll
            for (int i = 0; i < 2; ++i) {
                const bf16* p = &Asc[(wr * 32 + i * 16 + arow) * BKK + kh];
#pragma unroll
                for (int q = 0; q < 8; ++q) { afrag[i][q] = p[q]; afrag[i][8 + q] = p[16 + q]; }
            }
        }
        // B fragments (lane&15 = N column; lane<16 -> K 0..7,16..23; lane>=16 -> K 8..15,24..31)
#pragma unroll
        for (int j = 0; j < 4; ++j) {
            v16bf bfrag;
            {
                const int bcol = wc * 64 + j * 16 + (lane & 15);
                const int kb   = (lane >> 4) << 3;
#pragma unroll
                for (int q = 0; q < 8; ++q) { bfrag[q] = Bsc[(kb + q) * BN + bcol]; bfrag[8 + q] = Bsc[(16 + kb + q) * BN + bcol]; }
            }
#pragma unroll
            for (int i = 0; i < 2; ++i) {
                acc[i][j] = __builtin_amdgcn_wmma_f32_16x16x32_bf16(
                    false, afrag[i], false, bfrag, (short)0, acc[i][j], false, false);
            }
        }
    }

    // Epilogue: C/D layout => lane: col = lane&15, rows (lane>=16 ? 8..15 : 0..7)
    const int ccol0 = col0 + wc * 64 + (lane & 15);
    const int crow0 = row0 + wr * 32 + ((lane >> 4) << 3);
#pragma unroll
    for (int j = 0; j < 4; ++j) {
        const int col = ccol0 + j * 16;
        const float bv = bias ? bias[col] : 0.0f;
#pragma unroll
        for (int i = 0; i < 2; ++i) {
#pragma unroll
            for (int r = 0; r < 8; ++r) {
                float v = acc[i][j][r] + bv;
                size_t off = (size_t)(crow0 + i * 16 + r) * Nc + col;
                if (Cf) Cf[off] = v;
                if (Cb) Cb[off] = (bf16)v;
            }
        }
    }
}

// ---------------------------------------------------------------------------
// h = h + relu(LayerNorm(tmp) * g + bt)   (tmp already includes linear bias)
// One 256-thread block per row of 1024.
// ---------------------------------------------------------------------------
__global__ __launch_bounds__(256)
void ln_relu_res_kernel(const float* __restrict__ tmp, const float* __restrict__ g,
                        const float* __restrict__ bt,
                        float* __restrict__ h, bf16* __restrict__ hb)
{
    const int row = blockIdx.x;
    const int tid = threadIdx.x;
    const size_t base = (size_t)row * DH;
    float u[4];
    float s = 0.f, ss = 0.f;
#pragma unroll
    for (int i = 0; i < 4; ++i) {
        float v = tmp[base + tid + i * 256];
        u[i] = v; s += v; ss += v * v;
    }
#pragma unroll
    for (int o = 16; o > 0; o >>= 1) { s += __shfl_down(s, o, 32); ss += __shfl_down(ss, o, 32); }
    __shared__ float rs[8], rss[8];
    __shared__ float smu, srstd;
    if ((tid & 31) == 0) { rs[tid >> 5] = s; rss[tid >> 5] = ss; }
    __syncthreads();
    if (tid == 0) {
        float S = 0.f, SS = 0.f;
        for (int i = 0; i < 8; ++i) { S += rs[i]; SS += rss[i]; }
        float mu  = S * (1.0f / DH);
        float var = SS * (1.0f / DH) - mu * mu;   // population variance (jnp.var)
        smu = mu; srstd = rsqrtf(var + 1e-5f);
    }
    __syncthreads();
    const float mu = smu, rstd = srstd;
#pragma unroll
    for (int i = 0; i < 4; ++i) {
        int c = tid + i * 256;
        float v = (u[i] - mu) * rstd * g[c] + bt[c];
        v = fmaxf(v, 0.0f);
        float hn = h[base + c] + v;
        h[base + c]  = hn;
        hb[base + c] = (bf16)hn;
    }
}

// ---------------------------------------------------------------------------
// Vector quantization: per 16-dim row, argmin over 512 codes (LDS codebook).
// stats[0..511] = counts, stats[512] = sum ||z_q - z||^2
// ---------------------------------------------------------------------------
__global__ __launch_bounds__(256)
void vq_kernel(const float* __restrict__ z, const float* __restrict__ codebook,
               float* __restrict__ yhat, bf16* __restrict__ yb,
               float* __restrict__ stats)
{
    __shared__ float cb[KCODE * EDIM];   // 32 KB
    for (int i = threadIdx.x; i < KCODE * EDIM; i += 256) cb[i] = codebook[i];
    __syncthreads();

    const int r = blockIdx.x * 256 + threadIdx.x;
    const size_t base = (size_t)r * EDIM;
    float zv[EDIM];
    float zz = 0.f;
#pragma unroll
    for (int j = 0; j < EDIM; ++j) { zv[j] = z[base + j]; zz = fmaf(zv[j], zv[j], zz); }

    int best = 0; float bestd = 3.4e38f;
    for (int k = 0; k < KCODE; ++k) {
        const float* c = &cb[k * EDIM];
        float dot = 0.f, cn = 0.f;
#pragma unroll
        for (int j = 0; j < EDIM; ++j) { dot = fmaf(zv[j], c[j], dot); cn = fmaf(c[j], c[j], cn); }
        float d = zz + cn - 2.0f * dot;
        if (d < bestd) { bestd = d; best = k; }   // strict < == first min (jnp.argmin)
    }

    float lsum = 0.f;
#pragma unroll
    for (int j = 0; j < EDIM; ++j) {
        float q = cb[best * EDIM + j];
        yhat[base + j] = q;            // straight-through forward value == z_q
        yb[base + j]   = (bf16)q;
        float df = q - zv[j];
        lsum = fmaf(df, df, lsum);
    }
    atomicAdd(&stats[best], 1.0f);
#pragma unroll
    for (int o = 16; o > 0; o >>= 1) lsum += __shfl_down(lsum, o, 32);
    if ((threadIdx.x & 31) == 0) atomicAdd(&stats[KCODE], lsum);
}

__global__ void vq_finalize_kernel(const float* __restrict__ stats,
                                   float* __restrict__ out_loss,
                                   float* __restrict__ out_perp)
{
    const int t = threadIdx.x;  // 512 threads
    float c = stats[t] * (1.0f / (float)NROWS_VQ);
    float e = c * logf(c + 1e-10f);
#pragma unroll
    for (int o = 16; o > 0; o >>= 1) e += __shfl_down(e, o, 32);
    __shared__ float red[16];
    if ((t & 31) == 0) red[t >> 5] = e;
    __syncthreads();
    if (t == 0) {
        float s = 0.f;
        for (int i = 0; i < 16; ++i) s += red[i];
        *out_perp = expf(-s);
        *out_loss = stats[KCODE] * (1.25f / ((float)NROWS_VQ * (float)EDIM));
    }
}

// ---------------------------------------------------------------------------
// Small elementwise kernels
// ---------------------------------------------------------------------------
__global__ void zero_kernel(float* p, int n) {
    int i = blockIdx.x * 256 + threadIdx.x;
    if (i < n) p[i] = 0.f;
}

__global__ void cvt_f32_bf16_kernel(const float* __restrict__ src, bf16* __restrict__ dst, int n) {
    int i = blockIdx.x * 256 + threadIdx.x;
    if (i < n) dst[i] = (bf16)src[i];
}

__global__ void prep_kernel(const float* __restrict__ x, const float* __restrict__ scale,
                            const float* __restrict__ shift, bf16* __restrict__ out) {
    int i = blockIdx.x * 256 + threadIdx.x;     // exact N*DIN grid
    int c = i & (DIN - 1);
    out[i] = (bf16)((x[i] - shift[c]) / scale[c]);
}

__global__ void out_scale_kernel(const float* __restrict__ t, const float* __restrict__ scale,
                                 const float* __restrict__ shift, float* __restrict__ xhat) {
    int i = blockIdx.x * 256 + threadIdx.x;     // exact N*DIN grid
    int c = i & (DIN - 1);
    xhat[i] = scale[c] * t[i] + shift[c];       // t already includes b_out
}

// ---------------------------------------------------------------------------
// Host side
// ---------------------------------------------------------------------------
struct Enc {
    const float* w_in;  const float* b_in;
    const float* w[4];  const float* b[4]; const float* g[4]; const float* bt[4];
    const float* w_out; const float* b_out;
};

// Handles both dict-insertion order and jax-sorted-keys order, detected by sizes.
static void parse_enc(void* const* din, const int* sizes, int base, Enc& e) {
    if (sizes[base] == 1024) {
        // sorted keys: b_in, b_out, [b, bt, g, w] x4, w_in, w_out
        e.b_in  = (const float*)din[base + 0];
        e.b_out = (const float*)din[base + 1];
        for (int i = 0; i < 4; ++i) {
            e.b[i]  = (const float*)din[base + 2 + 4 * i + 0];
            e.bt[i] = (const float*)din[base + 2 + 4 * i + 1];
            e.g[i]  = (const float*)din[base + 2 + 4 * i + 2];
            e.w[i]  = (const float*)din[base + 2 + 4 * i + 3];
        }
        e.w_in  = (const float*)din[base + 16];
        e.w_out = (const float*)din[base + 17];
    } else {
        // insertion order: w_in, b_in, [w, b, g, bt] x4, w_out, b_out
        e.w_in = (const float*)din[base + 0];
        e.b_in = (const float*)din[base + 1];
        for (int i = 0; i < 4; ++i) {
            e.w[i]  = (const float*)din[base + 2 + 4 * i + 0];
            e.b[i]  = (const float*)din[base + 2 + 4 * i + 1];
            e.g[i]  = (const float*)din[base + 2 + 4 * i + 2];
            e.bt[i] = (const float*)din[base + 2 + 4 * i + 3];
        }
        e.w_out = (const float*)din[base + 16];
        e.b_out = (const float*)din[base + 17];
    }
}

extern "C" void kernel_launch(void* const* d_in, const int* in_sizes, int n_in,
                              void* d_out, int out_size, void* d_ws, size_t ws_size,
                              hipStream_t stream)
{
    (void)n_in; (void)out_size; (void)ws_size;

    const float* x        = (const float*)d_in[0];
    const float* scale    = (const float*)d_in[1];
    const float* shiftp   = (const float*)d_in[2];
    const float* codebook = (const float*)d_in[3];
    Enc ga, gs;
    parse_enc(d_in, in_sizes, 4, ga);
    parse_enc(d_in, in_sizes, 22, gs);

    // Workspace carve-out
    char* p = (char*)d_ws;
    auto carve = [&](size_t bytes) { char* r = p; p += (bytes + 255) & ~(size_t)255; return r; };

    const size_t wsz[6] = { (size_t)DIN * DH, (size_t)DH * DH, (size_t)DH * DH,
                            (size_t)DH * DH,  (size_t)DH * DH, (size_t)DH * MMDIM };
    bf16* ga_wb[6]; bf16* gs_wb[6];
    for (int i = 0; i < 6; ++i) ga_wb[i] = (bf16*)carve(wsz[i] * 2);
    for (int i = 0; i < 6; ++i) gs_wb[i] = (bf16*)carve(wsz[i] * 2);
    bf16*  inbf  = (bf16*) carve((size_t)NN * DIN * 2);   // encoder input (bf16)
    float* h     = (float*)carve((size_t)NN * DH * 4);    // f32 master activations
    bf16*  hb    = (bf16*) carve((size_t)NN * DH * 2);    // bf16 mirror for WMMA
    float* tmp   = (float*)carve((size_t)NN * DH * 4);    // GEMM outputs
    float* ybuf  = (float*)carve((size_t)NN * MMDIM * 4); // encoder output z
    float* stats = (float*)carve((KCODE + 1) * 4);        // counts + loss sum

    float* out      = (float*)d_out;
    float* out_loss = out;                                  // [0]
    float* out_xhat = out + 1;                              // [1 .. 1+N*DIN)
    float* out_perp = out + 1 + (size_t)NN * DIN;           // [1+N*DIN]
    float* out_yhat = out + 2 + (size_t)NN * DIN;           // [2+N*DIN ..)

    // Convert weights to bf16
    const float* ga_w[6] = { ga.w_in, ga.w[0], ga.w[1], ga.w[2], ga.w[3], ga.w_out };
    const float* gs_w[6] = { gs.w_in, gs.w[0], gs.w[1], gs.w[2], gs.w[3], gs.w_out };
    for (int i = 0; i < 6; ++i) {
        int n = (int)wsz[i];
        cvt_f32_bf16_kernel<<<(n + 255) / 256, 256, 0, stream>>>(ga_w[i], ga_wb[i], n);
        cvt_f32_bf16_kernel<<<(n + 255) / 256, 256, 0, stream>>>(gs_w[i], gs_wb[i], n);
    }

    // x_shift -> bf16
    prep_kernel<<<(NN * DIN) / 256, 256, 0, stream>>>(x, scale, shiftp, inbf);

    dim3 blk(256);
    dim3 g_wide(DH / BN, NN / BM);      // (8, 256)  : Nc = 1024
    dim3 g_narrow(MMDIM / BN, NN / BM); // (1, 256)  : Nc = 128

    // ---- encoder ga ----
    gemm_bf16_wmma_kernel<<<g_wide, blk, 0, stream>>>(inbf, ga_wb[0], ga.b_in, h, hb, NN, DH, DIN);
    for (int i = 0; i < 4; ++i) {
        gemm_bf16_wmma_kernel<<<g_wide, blk, 0, stream>>>(hb, ga_wb[1 + i], ga.b[i], tmp, (bf16*)nullptr, NN, DH, DH);
        ln_relu_res_kernel<<<NN, blk, 0, stream>>>(tmp, ga.g[i], ga.bt[i], h, hb);
    }
    gemm_bf16_wmma_kernel<<<g_narrow, blk, 0, stream>>>(hb, ga_wb[5], ga.b_out, ybuf, (bf16*)nullptr, NN, MMDIM, DH);

    // ---- vector quantization ----
    zero_kernel<<<3, 256, 0, stream>>>(stats, KCODE + 1);
    vq_kernel<<<NROWS_VQ / 256, blk, 0, stream>>>(ybuf, codebook, out_yhat, inbf, stats);

    // ---- decoder gs (input: quantized bf16 in inbf) ----
    gemm_bf16_wmma_kernel<<<g_wide, blk, 0, stream>>>(inbf, gs_wb[0], gs.b_in, h, hb, NN, DH, MMDIM);
    for (int i = 0; i < 4; ++i) {
        gemm_bf16_wmma_kernel<<<g_wide, blk, 0, stream>>>(hb, gs_wb[1 + i], gs.b[i], tmp, (bf16*)nullptr, NN, DH, DH);
        ln_relu_res_kernel<<<NN, blk, 0, stream>>>(tmp, gs.g[i], gs.bt[i], h, hb);
    }
    gemm_bf16_wmma_kernel<<<g_narrow, blk, 0, stream>>>(hb, gs_wb[5], gs.b_out, tmp, (bf16*)nullptr, NN, DIN, DH);

    out_scale_kernel<<<(NN * DIN) / 256, blk, 0, stream>>>(tmp, scale, shiftp, out_xhat);
    vq_finalize_kernel<<<1, 512, 0, stream>>>(stats, out_loss, out_perp);
}
